// QuantumHybridTransformer_65481071408416
// MI455X (gfx1250) — matrix operand order repaired and checked by reading, not compiled, not run on hardware
//
#include <hip/hip_runtime.h>

// ---------------- problem constants (match reference) ----------------
#define Bz   4
#define Sz   2048
#define Ez   512
#define Hz   8
#define DKz  64
#define Lz   2
#define NBz  4
#define FFz  2048
#define Cz   4
#define BSz  (Bz * Sz)          // 8192 rows

typedef __bf16 bf16;
typedef __attribute__((ext_vector_type(16))) __bf16 v16bf;
typedef __attribute__((ext_vector_type(8)))  float  v8f;
typedef __attribute__((ext_vector_type(4)))  float  f32x4;

#define WMMA_BF16(a, b, c) \
  __builtin_amdgcn_wmma_f32_16x16x32_bf16(false, (a), false, (b), (short)0, (c), false, false)

static __device__ __forceinline__ v16bf load_frag(const bf16* p0, const bf16* p1) {
  v16bf f;
  ((f32x4*)&f)[0] = *(const f32x4*)p0;   // 8 bf16 = 16B
  ((f32x4*)&f)[1] = *(const f32x4*)p1;   // 8 bf16 = 16B
  return f;
}

static __device__ __forceinline__ v8f v8f_zero() {
  v8f z = {0.f, 0.f, 0.f, 0.f, 0.f, 0.f, 0.f, 0.f};
  return z;
}

// CDNA5 async global->LDS copy, 16 bytes per lane (tracked by ASYNCcnt)
static __device__ __forceinline__ void async_cp16(unsigned lds_off, const bf16* gptr) {
  asm volatile("global_load_async_to_lds_b128 %0, %1, off"
               :: "v"(lds_off), "v"(gptr) : "memory");
}

// ---------------- weight convert + transpose: W[K,N] f32 -> Wt[N,K] bf16 ----
__global__ void k_convT(const float* __restrict__ W, bf16* __restrict__ Wt, int K, int N) {
  size_t base = (size_t)blockIdx.y * (size_t)K * (size_t)N;
  int tid = blockIdx.x * blockDim.x + threadIdx.x;
  if (tid >= K * N) return;
  int n = tid / K, kk = tid % K;
  Wt[base + tid] = (bf16)W[base + (size_t)kk * N + n];
}

// ---------------- embedding + sinusoidal positional encoding ---------------
__global__ void k_embed(const int* __restrict__ tok, const float* __restrict__ emb,
                        float* __restrict__ xf, bf16* __restrict__ xb) {
  size_t tid = (size_t)blockIdx.x * blockDim.x + threadIdx.x;
  if (tid >= (size_t)BSz * Ez) return;
  int e  = (int)(tid % Ez);
  size_t bs = tid / Ez;
  int s  = (int)(bs % Sz);
  int t  = tok[bs];
  float val = emb[(size_t)t * Ez + e];
  float ang = (float)s * __expf(-(float)(e & ~1) * (9.2103404f / (float)Ez));
  val += (e & 1) ? __cosf(ang) : __sinf(ang);
  xf[tid] = val;
  xb[tid] = (bf16)val;
}

// ======================== big WMMA GEMM =====================================
// C[M,N] = A[M,K] x Bt[N,K]^T, bf16 in / f32 accum.
// Workgroup: 256 thr = 8 waves as 2(M) x 4(N); block tile 128x256, BK=32.
// A/B panels double-buffered in LDS via global_load_async_to_lds_b128.
#define BM  128
#define BN  256
#define BK  32
#define PAD 40          // padded LDS row (bf16) -> 80B, conflict-free b128 reads

__global__ __launch_bounds__(256) void k_gemm_big(
    const bf16* __restrict__ A, const bf16* __restrict__ Bt,
    const float* __restrict__ bias, float* __restrict__ Cf, bf16* __restrict__ Cb,
    int M, int N, int K, int lda, int ldb, int ldc, int relu) {
  __shared__ bf16 sA[2 * BM * PAD];   // 20 KB
  __shared__ bf16 sB[2 * BN * PAD];   // 40 KB
  const int tid  = threadIdx.x;
  const int lane = tid & 31;
  const int hl   = lane >> 4;
  const int r    = lane & 15;
  const int wave = tid >> 5;
  const int wm   = wave >> 2;          // 0..1
  const int wn   = wave & 3;           // 0..3
  const int m0   = blockIdx.y * BM;
  const int n0   = blockIdx.x * BN;

  const bf16* gA = A  + (size_t)(m0 + (tid >> 1)) * lda + (tid & 1) * 16;
  const bf16* gB = Bt + (size_t)(n0 + tid) * ldb;
  const unsigned lA = (unsigned)(size_t)sA + (unsigned)(((tid >> 1) * PAD + (tid & 1) * 16) * 2);
  const unsigned lB = (unsigned)(size_t)sB + (unsigned)((tid * PAD) * 2);

  auto issue = [&](int s, int buf) {
    const bf16* ga = gA + s * BK;
    unsigned la = lA + (unsigned)(buf * BM * PAD * 2);
    async_cp16(la,      ga);
    async_cp16(la + 16, ga + 8);
    const bf16* gb = gB + s * BK;
    unsigned lb = lB + (unsigned)(buf * BN * PAD * 2);
    async_cp16(lb,      gb);
    async_cp16(lb + 16, gb + 8);
    async_cp16(lb + 32, gb + 16);
    async_cp16(lb + 48, gb + 24);
  };

  v8f acc[4][4];
#pragma unroll
  for (int mt = 0; mt < 4; ++mt)
#pragma unroll
    for (int nt = 0; nt < 4; ++nt) acc[mt][nt] = v8f_zero();

  const int nsteps = K / BK;
  issue(0, 0);
  for (int s = 0; s < nsteps; ++s) {
    const int cur = s & 1;
    if (s + 1 < nsteps) {
      issue(s + 1, cur ^ 1);
      asm volatile("s_wait_asynccnt 6" ::: "memory");
    } else {
      asm volatile("s_wait_asynccnt 0" ::: "memory");
    }
    __syncthreads();

    v16bf af[4], bfr[4];
#pragma unroll
    for (int mt = 0; mt < 4; ++mt) {
      const bf16* p = sA + cur * (BM * PAD) + (wm * 64 + mt * 16 + r) * PAD + hl * 8;
      af[mt] = load_frag(p, p + 16);
    }
#pragma unroll
    for (int nt = 0; nt < 4; ++nt) {
      const bf16* p = sB + cur * (BN * PAD) + (wn * 64 + nt * 16 + r) * PAD + hl * 8;
      bfr[nt] = load_frag(p, p + 16);
    }
#pragma unroll
    for (int mt = 0; mt < 4; ++mt)
#pragma unroll
      for (int nt = 0; nt < 4; ++nt)
        acc[mt][nt] = WMMA_BF16(af[mt], bfr[nt], acc[mt][nt]);

    __syncthreads();
  }

  float bv[4];
#pragma unroll
  for (int nt = 0; nt < 4; ++nt) {
    int n = n0 + wn * 64 + nt * 16 + r;
    bv[nt] = bias ? bias[n] : 0.f;
  }
#pragma unroll
  for (int mt = 0; mt < 4; ++mt) {
    size_t mbase = (size_t)(m0 + wm * 64 + mt * 16 + 8 * hl);
#pragma unroll
    for (int i = 0; i < 8; ++i) {
      size_t rowoff = (mbase + i) * ldc;
#pragma unroll
      for (int nt = 0; nt < 4; ++nt) {
        float v = acc[mt][nt][i] + bv[nt];
        if (relu) v = fmaxf(v, 0.f);
        size_t idx = rowoff + n0 + wn * 64 + nt * 16 + r;
        if (Cf) Cf[idx] = v;
        if (Cb) Cb[idx] = (bf16)v;
      }
    }
  }
}

// ---------------- small WMMA GEMM (quantum per-head 64x64 linears) ----------
__global__ __launch_bounds__(256) void k_gemm(
    const bf16* __restrict__ A, const bf16* __restrict__ Bt,
    const float* __restrict__ bias, float* __restrict__ Cf, bf16* __restrict__ Cb,
    int M, int N, int K, int lda, int ldb, int ldc, int relu,
    long long sAy, long long sBy, long long sBiy, long long sCy) {
  const int wave = threadIdx.x >> 5;
  const int lane = threadIdx.x & 31;
  const int hl   = lane >> 4;
  const int r    = lane & 15;
  int ntn  = N >> 4;
  int tile = blockIdx.x * 8 + wave;
  if (tile >= (M >> 4) * ntn) return;
  int tm = tile / ntn, tn = tile % ntn;

  const bf16*  Ay  = A  + (long long)blockIdx.y * sAy;
  const bf16*  By  = Bt + (long long)blockIdx.y * sBy;
  const float* biy = bias ? bias + (long long)blockIdx.y * sBiy : nullptr;
  long long    cy  = (long long)blockIdx.y * sCy;

  const bf16* arow = Ay + (size_t)(tm * 16 + r) * lda + hl * 8;
  const bf16* brow = By + (size_t)(tn * 16 + r) * ldb + hl * 8;

  v8f acc = v8f_zero();
  for (int kk = 0; kk < K; kk += 32) {
    v16bf af = load_frag(arow + kk, arow + kk + 16);
    v16bf bf = load_frag(brow + kk, brow + kk + 16);
    acc = WMMA_BF16(af, bf, acc);
  }

  float bv = biy ? biy[tn * 16 + r] : 0.f;
  size_t cbase = (size_t)(tm * 16 + 8 * hl) * ldc + tn * 16 + r;
#pragma unroll
  for (int i = 0; i < 8; ++i) {
    float v = acc[i] + bv;
    if (relu) v = fmaxf(v, 0.f);
    size_t idx = cbase + (size_t)i * ldc;
    if (Cf) Cf[cy + idx] = v;
    if (Cb) Cb[cy + idx] = (bf16)v;
  }
}

// ---------------- V transpose: [B,S,H*DK] -> [B,H,DK,S] ---------------------
__global__ void k_transpose_v(const bf16* __restrict__ v, bf16* __restrict__ vT) {
  size_t tid = (size_t)blockIdx.x * blockDim.x + threadIdx.x;
  if (tid >= (size_t)BSz * Ez) return;
  int e  = (int)(tid % Ez);
  size_t bs = tid / Ez;
  int s  = (int)(bs % Sz);
  int b  = (int)(bs / Sz);
  int h  = e / DKz, d = e % DKz;
  vT[((size_t)(b * Hz + h) * DKz + d) * Sz + s] = v[tid];
}

// ======================== flash attention ===================================
// One wave = 16 query rows; all 8 waves of a block share (b,h), so K/V 32-col
// chunks are staged cooperatively into LDS with async copies (double-buffered).
#define KPAD 72   // K-chunk LDS row (bf16): 144B, conflict-free b128 reads
#define VPAD 40   // V-chunk LDS row (bf16): 80B

__global__ __launch_bounds__(256) void k_attn(
    const bf16* __restrict__ q, const bf16* __restrict__ k,
    const bf16* __restrict__ vT, bf16* __restrict__ o) {
  __shared__ bf16 sK[2 * 32 * KPAD];   // 9 KB  : 32 seq rows x 64 dk
  __shared__ bf16 sV[2 * 64 * VPAD];   // 10 KB : 64 dk rows x 32 seq
  __shared__ bf16 sP[8][16 * 32];      // 8 KB  : per-wave P repack tile
  const int tid  = threadIdx.x;
  const int wave = tid >> 5;
  const int lane = tid & 31;
  const int hl   = lane >> 4;
  const int r    = lane & 15;
  const int bh = blockIdx.y, b = bh / Hz, h = bh % Hz;
  const int q0 = (blockIdx.x * 8 + wave) * 16;

  const bf16* qbase = q  + ((size_t)b * Sz + q0) * Ez + h * DKz;
  const bf16* kbase = k  + (size_t)b * Sz * Ez + h * DKz;   // [S][64]
  const bf16* vbase = vT + (size_t)(b * Hz + h) * DKz * Sz; // [64][S]

  // async staging: K chunk 32 rows x 128B -> thread t: row t>>3, chunk t&7
  //                V chunk 64 rows x 64B  -> thread t: row t>>2, chunk t&3
  const bf16* gK = kbase + (size_t)(tid >> 3) * Ez + (tid & 7) * 8;
  const bf16* gV = vbase + (size_t)(tid >> 2) * Sz + (tid & 3) * 8;
  const unsigned lK = (unsigned)(size_t)sK + (unsigned)(((tid >> 3) * KPAD + (tid & 7) * 8) * 2);
  const unsigned lV = (unsigned)(size_t)sV + (unsigned)(((tid >> 2) * VPAD + (tid & 3) * 8) * 2);

  auto issue = [&](int j0, int buf) {
    async_cp16(lK + (unsigned)(buf * 32 * KPAD * 2), gK + (size_t)j0 * Ez);
    async_cp16(lV + (unsigned)(buf * 64 * VPAD * 2), gV + j0);
  };

  const bf16* qr = qbase + (size_t)r * Ez;
  v16bf qa0 = load_frag(qr + hl * 8,      qr + 16 + hl * 8);
  v16bf qa1 = load_frag(qr + 32 + hl * 8, qr + 48 + hl * 8);

  float m[8], l[8];
  v8f acc0 = v8f_zero(), acc1 = v8f_zero(), acc2 = v8f_zero(), acc3 = v8f_zero();
#pragma unroll
  for (int i = 0; i < 8; ++i) { m[i] = -1e30f; l[i] = 0.f; }

  issue(0, 0);
  for (int j0 = 0, c = 0; j0 < Sz; j0 += 32, ++c) {
    const int cur = c & 1;
    if (j0 + 32 < Sz) {
      issue(j0 + 32, cur ^ 1);
      asm volatile("s_wait_asynccnt 2" ::: "memory");
    } else {
      asm volatile("s_wait_asynccnt 0" ::: "memory");
    }
    __syncthreads();                   // staged K/V visible to all waves

    const bf16* Kc = sK + cur * (32 * KPAD);
    const bf16* Vc = sV + cur * (64 * VPAD);

    // scores: two 16x16 tiles over K=DK=64
    v8f s0 = v8f_zero(), s1 = v8f_zero();
    const bf16* kp0 = Kc + r * KPAD + hl * 8;
    const bf16* kp1 = Kc + (16 + r) * KPAD + hl * 8;
    v16bf kb;
    kb = load_frag(kp0,      kp0 + 16); s0 = WMMA_BF16(qa0, kb, s0);
    kb = load_frag(kp0 + 32, kp0 + 48); s0 = WMMA_BF16(qa1, kb, s0);
    kb = load_frag(kp1,      kp1 + 16); s1 = WMMA_BF16(qa0, kb, s1);
    kb = load_frag(kp1 + 32, kp1 + 48); s1 = WMMA_BF16(qa1, kb, s1);

    // online softmax; C layout: element i -> row (i + 8*hl), col r
    float scl[8];
#pragma unroll
    for (int i = 0; i < 8; ++i) {
      float a0 = s0[i] * 0.125f;
      float a1 = s1[i] * 0.125f;
      float mx = fmaxf(a0, a1);
#pragma unroll
      for (int off = 8; off; off >>= 1) mx = fmaxf(mx, __shfl_xor(mx, off, 16));
      float mn = fmaxf(m[i], mx);
      float sl = __expf(m[i] - mn);
      float p0 = __expf(a0 - mn);
      float p1 = __expf(a1 - mn);
      float rs = p0 + p1;
#pragma unroll
      for (int off = 8; off; off >>= 1) rs += __shfl_xor(rs, off, 16);
      l[i] = l[i] * sl + rs;
      m[i] = mn;
      scl[i] = sl;
      s0[i] = p0;
      s1[i] = p1;
    }
#pragma unroll
    for (int i = 0; i < 8; ++i) {
      acc0[i] *= scl[i]; acc1[i] *= scl[i]; acc2[i] *= scl[i]; acc3[i] *= scl[i];
    }

    // repack P (C layout f32) -> A-frag (bf16 16x32) via per-wave LDS
    bf16* L = &sP[wave][0];
#pragma unroll
    for (int i = 0; i < 8; ++i) {
      int mm = i + 8 * hl;
      L[mm * 32 + r]      = (bf16)s0[i];
      L[mm * 32 + 16 + r] = (bf16)s1[i];
    }
    asm volatile("s_wait_dscnt 0" ::: "memory");
    v16bf pa = load_frag(L + r * 32 + hl * 8, L + r * 32 + 16 + hl * 8);

    // O += P (16x32) x V (32x64)
    const bf16* vp = Vc + r * VPAD + hl * 8;
    v16bf vb;
    vb = load_frag(vp,             vp + 16);             acc0 = WMMA_BF16(pa, vb, acc0);
    vb = load_frag(vp + 16 * VPAD, vp + 16 * VPAD + 16); acc1 = WMMA_BF16(pa, vb, acc1);
    vb = load_frag(vp + 32 * VPAD, vp + 32 * VPAD + 16); acc2 = WMMA_BF16(pa, vb, acc2);
    vb = load_frag(vp + 48 * VPAD, vp + 48 * VPAD + 16); acc3 = WMMA_BF16(pa, vb, acc3);

    __syncthreads();                   // all waves done with this buffer
  }

  bf16* ob = o + ((size_t)b * Sz + q0) * Ez + h * DKz;
#pragma unroll
  for (int i = 0; i < 8; ++i) {
    float inv = 1.f / l[i];
    size_t row = (size_t)(i + 8 * hl) * Ez + r;
    ob[row]      = (bf16)(acc0[i] * inv);
    ob[row + 16] = (bf16)(acc1[i] * inv);
    ob[row + 32] = (bf16)(acc2[i] * inv);
    ob[row + 48] = (bf16)(acc3[i] * inv);
  }
}

// ---------------- residual add + layernorm (one wave per row) ---------------
__global__ __launch_bounds__(256) void k_ln(
    const float* __restrict__ xin, const float* __restrict__ res,
    const float* __restrict__ g, const float* __restrict__ bta,
    float* __restrict__ xout, bf16* __restrict__ xb) {
  int wave = threadIdx.x >> 5, lane = threadIdx.x & 31;
  int row = blockIdx.x * 8 + wave;
  const float* xi = xin + (size_t)row * Ez;
  const float* rs = res + (size_t)row * Ez;
  float v[16];
  float s = 0.f;
#pragma unroll
  for (int j = 0; j < 16; ++j) { v[j] = xi[lane + 32 * j] + rs[lane + 32 * j]; s += v[j]; }
#pragma unroll
  for (int off = 16; off; off >>= 1) s += __shfl_xor(s, off, 32);
  float mean = s * (1.f / Ez);
  float ss = 0.f;
#pragma unroll
  for (int j = 0; j < 16; ++j) { float d = v[j] - mean; ss += d * d; }
#pragma unroll
  for (int off = 16; off; off >>= 1) ss += __shfl_xor(ss, off, 32);
  float rstd = rsqrtf(ss * (1.f / Ez) + 1e-5f);
  float* xo = xout + (size_t)row * Ez;
  bf16*  xc = xb + (size_t)row * Ez;
#pragma unroll
  for (int j = 0; j < 16; ++j) {
    int e = lane + 32 * j;
    float o = (v[j] - mean) * rstd * g[e] + bta[e];
    xo[e] = o;
    xc[e] = (bf16)o;
  }
}

// ---------------- mean pool (two-phase, deterministic) + classifier ---------
#define PSLAB 16
__global__ void k_pool1(const float* __restrict__ xf, float* __restrict__ part) {
  int b = blockIdx.x, slab = blockIdx.y;
  int t0 = slab * (Sz / PSLAB);
  for (int e = threadIdx.x; e < Ez; e += blockDim.x) {
    float s = 0.f;
    for (int t = t0; t < t0 + Sz / PSLAB; ++t) s += xf[((size_t)b * Sz + t) * Ez + e];
    part[((size_t)b * PSLAB + slab) * Ez + e] = s;
  }
}
__global__ void k_pool2(const float* __restrict__ part, float* __restrict__ pooled) {
  int b = blockIdx.x;
  for (int e = threadIdx.x; e < Ez; e += blockDim.x) {
    float s = 0.f;
    for (int j = 0; j < PSLAB; ++j) s += part[((size_t)b * PSLAB + j) * Ez + e];
    pooled[b * Ez + e] = s * (1.f / Sz);
  }
}

__global__ __launch_bounds__(256) void k_cls(
    const float* __restrict__ pooled, const float* __restrict__ wcls,
    const float* __restrict__ bcls, float* __restrict__ out) {
  __shared__ float red[256];
  int idx = blockIdx.x, b = idx / Cz, c = idx % Cz;
  float s = 0.f;
  for (int e = threadIdx.x; e < Ez; e += 256) s += pooled[b * Ez + e] * wcls[e * Cz + c];
  red[threadIdx.x] = s;
  __syncthreads();
  for (int off = 128; off; off >>= 1) {
    if (threadIdx.x < off) red[threadIdx.x] += red[threadIdx.x + off];
    __syncthreads();
  }
  if (threadIdx.x == 0) out[idx] = red[0] + bcls[c];
}

// ---------------------------------------------------------------------------
extern "C" void kernel_launch(void* const* d_in, const int* in_sizes, int n_in,
                              void* d_out, int out_size, void* d_ws, size_t ws_size,
                              hipStream_t stream) {
  const int*   tokens = (const int*)d_in[0];
  const float* emb    = (const float*)d_in[1];
  const float* Wq     = (const float*)d_in[2];
  const float* Wk     = (const float*)d_in[3];
  const float* Wv     = (const float*)d_in[4];
  const float* Wo     = (const float*)d_in[5];
  const float* qh_w   = (const float*)d_in[6];
  const float* qh_b   = (const float*)d_in[7];
  const float* kh_w   = (const float*)d_in[8];
  const float* kh_b   = (const float*)d_in[9];
  const float* vh_w   = (const float*)d_in[10];
  const float* vh_b   = (const float*)d_in[11];
  const float* ln1_g  = (const float*)d_in[12];
  const float* ln1_b  = (const float*)d_in[13];
  const float* ln2_g  = (const float*)d_in[14];
  const float* ln2_b  = (const float*)d_in[15];
  const float* f_w1   = (const float*)d_in[16];
  const float* f_b1   = (const float*)d_in[17];
  const float* f_w2   = (const float*)d_in[18];
  const float* f_b2   = (const float*)d_in[19];
  const float* cls_w  = (const float*)d_in[20];
  const float* cls_b  = (const float*)d_in[21];
  float* out = (float*)d_out;

  char* w = (char*)d_ws;
  auto carve = [&](size_t bytes) -> void* {
    void* p = (void*)w;
    w += (bytes + 255) & ~(size_t)255;
    return p;
  };
  const size_t actE = (size_t)BSz * Ez;
  float* xf    = (float*)carve(actE * 4);
  bf16*  xb    = (bf16*) carve(actE * 2);
  bf16*  qb    = (bf16*) carve(actE * 2);
  bf16*  kb    = (bf16*) carve(actE * 2);
  bf16*  vb    = (bf16*) carve(actE * 2);
  bf16*  tmpb  = (bf16*) carve(actE * 2);
  bf16*  vTb   = (bf16*) carve(actE * 2);
  bf16*  attnb = (bf16*) carve(actE * 2);
  float* projf = (float*)carve(actE * 4);
  bf16*  ffnh  = (bf16*) carve((size_t)BSz * FFz * 2);
  bf16*  WqT   = (bf16*) carve((size_t)NBz * Ez * Ez * 2);
  bf16*  WkT   = (bf16*) carve((size_t)NBz * Ez * Ez * 2);
  bf16*  WvT   = (bf16*) carve((size_t)NBz * Ez * Ez * 2);
  bf16*  WoT   = (bf16*) carve((size_t)NBz * Ez * Ez * 2);
  bf16*  qhT   = (bf16*) carve((size_t)NBz * Lz * Hz * DKz * DKz * 2);
  bf16*  khT   = (bf16*) carve((size_t)NBz * Lz * Hz * DKz * DKz * 2);
  bf16*  vhT   = (bf16*) carve((size_t)NBz * Lz * Hz * DKz * DKz * 2);
  bf16*  f1T   = (bf16*) carve((size_t)NBz * Ez * FFz * 2);
  bf16*  f2T   = (bf16*) carve((size_t)NBz * FFz * Ez * 2);
  float* part   = (float*)carve((size_t)Bz * PSLAB * Ez * 4);
  float* pooled = (float*)carve((size_t)Bz * Ez * 4);
  (void)ws_size; (void)in_sizes; (void)n_in; (void)out_size;

  k_convT<<<dim3((Ez * Ez + 255) / 256, NBz), 256, 0, stream>>>(Wq, WqT, Ez, Ez);
  k_convT<<<dim3((Ez * Ez + 255) / 256, NBz), 256, 0, stream>>>(Wk, WkT, Ez, Ez);
  k_convT<<<dim3((Ez * Ez + 255) / 256, NBz), 256, 0, stream>>>(Wv, WvT, Ez, Ez);
  k_convT<<<dim3((Ez * Ez + 255) / 256, NBz), 256, 0, stream>>>(Wo, WoT, Ez, Ez);
  k_convT<<<dim3((DKz * DKz + 255) / 256, NBz * Lz * Hz), 256, 0, stream>>>(qh_w, qhT, DKz, DKz);
  k_convT<<<dim3((DKz * DKz + 255) / 256, NBz * Lz * Hz), 256, 0, stream>>>(kh_w, khT, DKz, DKz);
  k_convT<<<dim3((DKz * DKz + 255) / 256, NBz * Lz * Hz), 256, 0, stream>>>(vh_w, vhT, DKz, DKz);
  k_convT<<<dim3((Ez * FFz + 255) / 256, NBz), 256, 0, stream>>>(f_w1, f1T, Ez, FFz);
  k_convT<<<dim3((FFz * Ez + 255) / 256, NBz), 256, 0, stream>>>(f_w2, f2T, FFz, Ez);

  k_embed<<<(int)((actE + 255) / 256), 256, 0, stream>>>(tokens, emb, xf, xb);

  auto gemmB = [&](const bf16* A, const bf16* Bt, const float* bias,
                   float* Cf, bf16* Cb, int M, int N, int K,
                   int lda, int ldb, int ldc, int relu) {
    dim3 g(N / BN, M / BM);
    k_gemm_big<<<g, 256, 0, stream>>>(A, Bt, bias, Cf, Cb, M, N, K, lda, ldb, ldc, relu);
  };
  auto gemmS = [&](const bf16* A, const bf16* Bt, const float* bias, bf16* Cb,
                   long long sAy, long long sBy, long long sBiy, long long sCy) {
    dim3 g(((BSz / 16) * (DKz / 16) + 7) / 8, Hz);
    k_gemm<<<g, 256, 0, stream>>>(A, Bt, bias, nullptr, Cb, BSz, DKz, DKz, Ez, DKz, Ez, 0,
                                  sAy, sBy, sBiy, sCy);
  };

  for (int i = 0; i < NBz; ++i) {
    gemmB(xb, WqT + (size_t)i * Ez * Ez, nullptr, nullptr, qb, BSz, Ez, Ez, Ez, Ez, Ez, 0);
    gemmB(xb, WkT + (size_t)i * Ez * Ez, nullptr, nullptr, kb, BSz, Ez, Ez, Ez, Ez, Ez, 0);
    gemmB(xb, WvT + (size_t)i * Ez * Ez, nullptr, nullptr, vb, BSz, Ez, Ez, Ez, Ez, Ez, 0);

    for (int l = 0; l < Lz; ++l) {
      size_t wo = (size_t)(i * Lz + l) * Hz * DKz * DKz;
      size_t bo = (size_t)(i * Lz + l) * Hz * DKz;
      bf16* srcQ = (l == 0) ? qb : tmpb;  bf16* dstQ = (l == 0) ? tmpb : qb;
      gemmS(srcQ, qhT + wo, qh_b + bo, dstQ, DKz, (long long)DKz * DKz, DKz, DKz);
      bf16* srcK = (l == 0) ? kb : tmpb;  bf16* dstK = (l == 0) ? tmpb : kb;
      gemmS(srcK, khT + wo, kh_b + bo, dstK, DKz, (long long)DKz * DKz, DKz, DKz);
      bf16* srcV = (l == 0) ? vb : tmpb;  bf16* dstV = (l == 0) ? tmpb : vb;
      gemmS(srcV, vhT + wo, vh_b + bo, dstV, DKz, (long long)DKz * DKz, DKz, DKz);
    }

    k_transpose_v<<<(int)((actE + 255) / 256), 256, 0, stream>>>(vb, vTb);
    k_attn<<<dim3(Sz / 128, Bz * Hz), 256, 0, stream>>>(qb, kb, vTb, attnb);

    gemmB(attnb, WoT + (size_t)i * Ez * Ez, nullptr, projf, nullptr, BSz, Ez, Ez, Ez, Ez, Ez, 0);
    k_ln<<<BSz / 8, 256, 0, stream>>>(xf, projf, ln1_g + i * Ez, ln1_b + i * Ez, xf, xb);

    gemmB(xb, f1T + (size_t)i * Ez * FFz, f_b1 + i * FFz, nullptr, ffnh,
          BSz, FFz, Ez, Ez, Ez, FFz, 1);
    gemmB(ffnh, f2T + (size_t)i * FFz * Ez, f_b2 + i * Ez, projf, nullptr,
          BSz, Ez, FFz, FFz, FFz, Ez, 0);
    k_ln<<<BSz / 8, 256, 0, stream>>>(xf, projf, ln2_g + i * Ez, ln2_b + i * Ez, xf, xb);
  }

  k_pool1<<<dim3(Bz, PSLAB), 256, 0, stream>>>(xf, part);
  k_pool2<<<Bz, 256, 0, stream>>>(part, pooled);
  k_cls<<<Bz * Cz, 256, 0, stream>>>(pooled, cls_w, cls_b, out);
}
